// TurkishAttention_6313601925407
// MI455X (gfx1250) — matrix operand order repaired and checked by reading, compiled, not run on hardware
//
#include <hip/hip_runtime.h>
#include <hip/hip_bf16.h>
#include <stdint.h>

// Problem constants (from reference)
#define B_   2
#define S_   2048
#define HID_ 3072
#define H_   16
#define KV_  8
#define D_   256
#define NTOK (B_*S_)   // 4096 tokens
#define QN   (H_*D_)   // 4096
#define KVN  (KV_*D_)  // 2048

typedef __attribute__((ext_vector_type(16))) __bf16 bf16x16;
typedef __attribute__((ext_vector_type(8)))  __bf16 bf16x8;
typedef __attribute__((ext_vector_type(8)))  float  f32x8;

__device__ __forceinline__ f32x8 wmma_bf16(bf16x16 a, bf16x16 b, f32x8 c) {
  // emits v_wmma_f32_16x16x32_bf16
  return __builtin_amdgcn_wmma_f32_16x16x32_bf16(false, a, false, b, (short)0, c, false, false);
}

// A-matrix fragment (16x32 bf16, M=lane%16):
//  lanes 0-15 hold K {0..15}, lanes 16-31 hold K {16..31}
__device__ __forceinline__ bf16x16 load_a_frag(const __bf16* p, int lh) {
  const __bf16* q = p + lh * 16;
  return *(const bf16x16*)q;
}

// ---------------------------------------------------------------------------
// f32 -> bf16 elementwise
__global__ __launch_bounds__(256,1) void f32_to_bf16_k(const float* __restrict__ x,
                                                       __bf16* __restrict__ y, int n) {
  int i = blockIdx.x * blockDim.x + threadIdx.x;
  if (i < n) y[i] = (__bf16)x[i];
}

// Transpose-convert: W[K][N] f32  ->  Wt[N][K] bf16  (B operand: contiguous K)
__global__ __launch_bounds__(256,1) void transpose_cvt_k(const float* __restrict__ W,
                                                         __bf16* __restrict__ Wt,
                                                         int K, int N) {
  __shared__ float tile[32][33];
  const int k0 = blockIdx.y * 32, n0 = blockIdx.x * 32;
  const int tx = threadIdx.x, ty = threadIdx.y;   // 32 x 8
  #pragma unroll
  for (int i = 0; i < 32; i += 8)
    tile[ty + i][tx] = W[(size_t)(k0 + ty + i) * N + n0 + tx];
  __syncthreads();
  #pragma unroll
  for (int i = 0; i < 32; i += 8)
    Wt[(size_t)(n0 + ty + i) * K + k0 + tx] = (__bf16)tile[tx][ty + i];
}

// ---------------------------------------------------------------------------
// Generic GEMM: C[M][N] f32 = A[M][K] bf16 @ Bt[N][K]^T bf16
// 8 waves/block: block tile 64x256, wave tile 32x64 (2 M-frags x 4 N-frags).
// B fragments reused across both M-frags: 12 b128 loads per 8 WMMAs.
__global__ __launch_bounds__(256,1) void gemm_bf16_k(const __bf16* __restrict__ A,
                                                     const __bf16* __restrict__ Bt,
                                                     float* __restrict__ C,
                                                     int M, int N, int K) {
  const int lane = threadIdx.x & 31;
  const int wave = threadIdx.x >> 5;
  const int lm = lane & 15, lh = lane >> 4;
  const int m0 = blockIdx.y * 64 + (wave >> 2) * 32;
  const int n0 = blockIdx.x * 256 + (wave & 3) * 64;

  f32x8 zero = {0,0,0,0,0,0,0,0};
  f32x8 acc[2][4];
  #pragma unroll
  for (int i = 0; i < 2; i++)
    #pragma unroll
    for (int t = 0; t < 4; t++) acc[i][t] = zero;

  const __bf16* ap0 = A  + (size_t)(m0 + lm) * K;
  const __bf16* ap1 = A  + (size_t)(m0 + 16 + lm) * K;
  const __bf16* bp  = Bt + (size_t)(n0 + lm) * K + lh * 16;

  for (int k = 0; k < K; k += 32) {
    bf16x16 a0 = load_a_frag(ap0 + k, lh);
    bf16x16 a1 = load_a_frag(ap1 + k, lh);
    #pragma unroll
    for (int t = 0; t < 4; t++) {
      bf16x16 b = *(const bf16x16*)(bp + (size_t)t * 16 * K + k);
      acc[0][t] = wmma_bf16(a0, b, acc[0][t]);
      acc[1][t] = wmma_bf16(a1, b, acc[1][t]);
    }
  }
  // C/D layout: VGPR r -> row (+ r + lh*8), lane -> column (n0 + t*16 + lm)
  #pragma unroll
  for (int i = 0; i < 2; i++) {
    #pragma unroll
    for (int t = 0; t < 4; t++) {
      float* cp = C + (size_t)(m0 + i * 16 + lh * 8) * N + n0 + t * 16 + lm;
      #pragma unroll
      for (int r = 0; r < 8; r++) cp[(size_t)r * N] = acc[i][t][r];
    }
  }
}

// ---------------------------------------------------------------------------
// RMSNorm (1+w) + RoPE (+ optional q-scale), writing head-major bf16:
//  Out[b][head][s][d].  One block = one (token, head), 256 threads = D lanes.
__global__ __launch_bounds__(256,1) void norm_rope_k(const float* __restrict__ X,
                                                     const float* __restrict__ w,
                                                     const float* __restrict__ cosT,
                                                     const float* __restrict__ sinT,
                                                     __bf16* __restrict__ Out,
                                                     int nheads, float scale) {
  const int tok = blockIdx.x, h = blockIdx.y, d = threadIdx.x;
  const int s = tok & (S_ - 1), b = tok >> 11;
  __shared__ float xn[D_];
  __shared__ float red[8];

  float x = X[((size_t)tok * nheads + h) * D_ + d];
  float ss = x * x;
  #pragma unroll
  for (int o = 16; o > 0; o >>= 1) ss += __shfl_xor(ss, o, 32);
  if ((threadIdx.x & 31) == 0) red[threadIdx.x >> 5] = ss;
  __syncthreads();
  float tot = 0.f;
  #pragma unroll
  for (int i = 0; i < 8; i++) tot += red[i];

  float n = x * rsqrtf(tot * (1.0f / D_) + 1e-6f) * (1.0f + w[d]);
  xn[d] = n;
  __syncthreads();

  float c  = cosT[s * (D_ / 2) + (d & (D_ / 2 - 1))];
  float sn = sinT[s * (D_ / 2) + (d & (D_ / 2 - 1))];
  float o;
  if (d < D_ / 2) o = n * c - xn[d + D_ / 2] * sn;
  else            o = xn[d - D_ / 2] * sn + n * c;

  Out[(((size_t)(b * nheads + h)) * S_ + s) * D_ + d] = (__bf16)(o * scale);
}

// V f32 [b][s][kv*D] -> Vt bf16 [b][kv][D][S]  (PV B-operand: contiguous keys)
__global__ __launch_bounds__(256,1) void v_transpose_k(const float* __restrict__ V,
                                                       __bf16* __restrict__ Vt) {
  __shared__ float tile[32][33];
  const int s0 = blockIdx.x * 32, d0 = blockIdx.y * 32, bk = blockIdx.z;
  const int b = bk >> 3, kv = bk & 7;
  const int tx = threadIdx.x, ty = threadIdx.y;  // 32 x 8
  #pragma unroll
  for (int i = 0; i < 32; i += 8)
    tile[ty + i][tx] = V[((size_t)(b * S_ + s0 + ty + i)) * KVN + kv * D_ + d0 + tx];
  __syncthreads();
  #pragma unroll
  for (int i = 0; i < 32; i += 8)
    Vt[((size_t)bk * D_ + d0 + ty + i) * S_ + s0 + tx] = (__bf16)tile[tx][ty + i];
}

// ---------------------------------------------------------------------------
// Flash attention: 4 waves/block, one wave = 16 q rows x full D=256.
// Q pre-scaled by D^-0.5 in norm_rope.  GQA: kv head = h >> 1.
__global__ __launch_bounds__(128,1) void flash_attn_k(const __bf16* __restrict__ Q,   // [B][H][S][D]
                                                      const __bf16* __restrict__ Kc,  // [B][KV][S][D]
                                                      const __bf16* __restrict__ Vt,  // [B][KV][D][S]
                                                      __bf16* __restrict__ O) {       // [B*S][H*D]
  const int lane = threadIdx.x & 31;
  const int wave = threadIdx.x >> 5;
  const int lm = lane & 15, lh = lane >> 4;
  const int b = blockIdx.z, h = blockIdx.y;
  const int kvh = h >> 1;
  const int q0 = blockIdx.x * 64 + wave * 16;

  const __bf16* Qp = Q  + (((size_t)(b * H_  + h  )) * S_ + q0) * D_;
  const __bf16* Kp = Kc + ((size_t)(b * KV_ + kvh)) * S_ * D_;
  const __bf16* Vp = Vt + ((size_t)(b * KV_ + kvh)) * D_ * S_;

  __shared__ __attribute__((aligned(64))) __bf16 Pl[4][16][32];

  // Resident Q fragments: 8 chunks of 16x32
  bf16x16 qf[8];
  #pragma unroll
  for (int c = 0; c < 8; c++) qf[c] = load_a_frag(Qp + (size_t)lm * D_ + c * 32, lh);

  f32x8 zero = {0,0,0,0,0,0,0,0};
  f32x8 o[16];
  #pragma unroll
  for (int t = 0; t < 16; t++) o[t] = zero;
  float m[8], l[8];
  #pragma unroll
  for (int r = 0; r < 8; r++) { m[r] = -1e30f; l[r] = 0.f; }

  for (int kb = 0; kb < q0 + 16; kb += 32) {
    // S = Q @ K^T : two 16x16 score tiles (keys kb..kb+15, kb+16..kb+31)
    f32x8 sa = zero, sb = zero;
    #pragma unroll
    for (int c = 0; c < 8; c++) {
      bf16x16 ka = *(const bf16x16*)(Kp + (size_t)(kb      + lm) * D_ + c * 32 + lh * 16);
      sa = wmma_bf16(qf[c], ka, sa);
    }
    #pragma unroll
    for (int c = 0; c < 8; c++) {
      bf16x16 kbv = *(const bf16x16*)(Kp + (size_t)(kb + 16 + lm) * D_ + c * 32 + lh * 16);
      sb = wmma_bf16(qf[c], kbv, sb);
    }

    // Causal mask + online softmax (row = r + lh*8, col = lm; half-wave reduce)
    const int ca = kb + lm, cb = kb + 16 + lm;
    #pragma unroll
    for (int r = 0; r < 8; r++) {
      const int qi = q0 + r + lh * 8;
      float va = (ca <= qi) ? sa[r] : -1e30f;
      float vb = (cb <= qi) ? sb[r] : -1e30f;
      float vm = fmaxf(va, vb);
      #pragma unroll
      for (int od = 1; od < 16; od <<= 1) vm = fmaxf(vm, __shfl_xor(vm, od, 32));
      float mn = fmaxf(m[r], vm);
      float alpha = __expf(m[r] - mn);
      float pa = __expf(va - mn);
      float pb = __expf(vb - mn);
      float rs = pa + pb;
      #pragma unroll
      for (int od = 1; od < 16; od <<= 1) rs += __shfl_xor(rs, od, 32);
      l[r] = l[r] * alpha + rs;
      m[r] = mn;
      #pragma unroll
      for (int t = 0; t < 16; t++) o[t][r] *= alpha;
      // bounce P through LDS to re-layout C-tile -> A-fragment
      Pl[wave][r + lh * 8][lm]      = (__bf16)pa;
      Pl[wave][r + lh * 8][16 + lm] = (__bf16)pb;
    }
    asm volatile("s_wait_dscnt 0x0" ::: "memory");

    // O += P(16x32) @ V(32xD)
    bf16x16 pf = load_a_frag(&Pl[wave][lm][0], lh);
    #pragma unroll
    for (int t = 0; t < 16; t++) {
      bf16x16 vv = *(const bf16x16*)(Vp + (size_t)(t * 16 + lm) * S_ + kb + lh * 16);
      o[t] = wmma_bf16(pf, vv, o[t]);
    }
  }

  // epilogue: divide by l, write token-major bf16 [b*S+s][h*D + d]
  #pragma unroll
  for (int r = 0; r < 8; r++) {
    float inv = 1.0f / l[r];
    const size_t row = (size_t)(b * S_ + q0 + r + lh * 8) * (size_t)QN + (size_t)h * D_;
    #pragma unroll
    for (int t = 0; t < 16; t++)
      O[row + t * 16 + lm] = (__bf16)(o[t][r] * inv);
  }
}

// ---------------------------------------------------------------------------
extern "C" void kernel_launch(void* const* d_in, const int* in_sizes, int n_in,
                              void* d_out, int out_size, void* d_ws, size_t ws_size,
                              hipStream_t stream) {
  const float* hs  = (const float*)d_in[0];
  const float* fc  = (const float*)d_in[1];
  const float* fs  = (const float*)d_in[2];
  /* d_in[3] = mask : causal handled in-kernel */
  const float* qw  = (const float*)d_in[4];
  const float* kw  = (const float*)d_in[5];
  const float* vw  = (const float*)d_in[6];
  const float* ow  = (const float*)d_in[7];
  const float* qnw = (const float*)d_in[8];
  const float* knw = (const float*)d_in[9];
  (void)in_sizes; (void)n_in; (void)out_size; (void)ws_size;

  char* p = (char*)d_ws;
  auto alloc = [&](size_t bytes) -> char* {
    char* r = p; p += (bytes + 255) & ~(size_t)255; return r;
  };
  __bf16* Xb  = (__bf16*)alloc((size_t)NTOK * HID_ * 2);   // hidden bf16
  __bf16* QWt = (__bf16*)alloc((size_t)QN   * HID_ * 2);   // q_w^T bf16
  __bf16* KWt = (__bf16*)alloc((size_t)KVN  * HID_ * 2);
  __bf16* VWt = (__bf16*)alloc((size_t)KVN  * HID_ * 2);
  __bf16* OWt = (__bf16*)alloc((size_t)HID_ * QN   * 2);   // o_w^T bf16
  float*  Qf  = (float*) alloc((size_t)NTOK * QN  * 4);
  float*  Kf  = (float*) alloc((size_t)NTOK * KVN * 4);
  float*  Vf  = (float*) alloc((size_t)NTOK * KVN * 4);
  __bf16* Qb  = (__bf16*)alloc((size_t)B_ * H_  * S_ * D_ * 2);
  __bf16* Kb  = (__bf16*)alloc((size_t)B_ * KV_ * S_ * D_ * 2);
  __bf16* Vtb = (__bf16*)alloc((size_t)B_ * KV_ * D_ * S_ * 2);
  __bf16* Ab  = (__bf16*)alloc((size_t)NTOK * QN * 2);

  // 1) hidden -> bf16
  { int n = NTOK * HID_;
    f32_to_bf16_k<<<(n + 255) / 256, 256, 0, stream>>>(hs, Xb, n); }

  // 2) weight transpose + convert (Bt[N][K] layouts)
  transpose_cvt_k<<<dim3(QN  / 32, HID_ / 32), dim3(32, 8), 0, stream>>>(qw, QWt, HID_, QN);
  transpose_cvt_k<<<dim3(KVN / 32, HID_ / 32), dim3(32, 8), 0, stream>>>(kw, KWt, HID_, KVN);
  transpose_cvt_k<<<dim3(KVN / 32, HID_ / 32), dim3(32, 8), 0, stream>>>(vw, VWt, HID_, KVN);
  transpose_cvt_k<<<dim3(HID_ / 32, QN / 32),  dim3(32, 8), 0, stream>>>(ow, OWt, QN, HID_);

  // 3) QKV projections (f32 accum out)  [block tile 64x256]
  gemm_bf16_k<<<dim3(QN  / 256, NTOK / 64), 256, 0, stream>>>(Xb, QWt, Qf, NTOK, QN,  HID_);
  gemm_bf16_k<<<dim3(KVN / 256, NTOK / 64), 256, 0, stream>>>(Xb, KWt, Kf, NTOK, KVN, HID_);
  gemm_bf16_k<<<dim3(KVN / 256, NTOK / 64), 256, 0, stream>>>(Xb, VWt, Vf, NTOK, KVN, HID_);

  // 4) RMSNorm + RoPE (+ SCALE folded into Q)
  norm_rope_k<<<dim3(NTOK, H_),  256, 0, stream>>>(Qf, qnw, fc, fs, Qb, H_,  0.0625f /* 256^-0.5 */);
  norm_rope_k<<<dim3(NTOK, KV_), 256, 0, stream>>>(Kf, knw, fc, fs, Kb, KV_, 1.0f);

  // 5) V -> V^T bf16
  v_transpose_k<<<dim3(S_ / 32, D_ / 32, B_ * KV_), dim3(32, 8), 0, stream>>>(Vf, Vtb);

  // 6) flash attention
  flash_attn_k<<<dim3(S_ / 64, H_, B_), 128, 0, stream>>>(Qb, Kb, Vtb, Ab);

  // 7) output projection -> f32 d_out
  gemm_bf16_k<<<dim3(HID_ / 256, NTOK / 64), 256, 0, stream>>>(Ab, OWt, (float*)d_out,
                                                               NTOK, HID_, QN);
}
